// FactoredBlock_13666585936404
// MI455X (gfx1250) — matrix operand order repaired and checked by reading, not compiled, not run on hardware
//
#include <hip/hip_runtime.h>
#include <stdint.h>

// ---------------------------------------------------------------------------
// FactoredBlock (NNUE-style factored input layer), fused for MI455X / gfx1250.
//
//   out[r,:] = sum_k values[r,k] * W[ f[active_idx[r,k]] , : ]
//
// Per 32-row tile: scatter-add into LDS f32 accumulator (ds_add_f32),
// convert to bf16 *directly in WMMA A-fragment order* ([frag][qword][lane]
// u64 interleave -> conflict-free ds_load_b64, immediate-offset addressing),
// then v_wmma_f32_16x16x32_bf16 against a bf16-transposed copy of the
// 768x256 weight matrix in d_ws (L2-resident, 384 KB). 4 independent WMMAs
// per K-step (2 row-blocks x 2 col-blocks) hide WMMA->WMMA spacing.
// HBM traffic ~47 MB -> ~2us floor at 23.3 TB/s; compute is negligible.
// ---------------------------------------------------------------------------

typedef __bf16 bf16_t;
typedef __attribute__((ext_vector_type(16))) __bf16 v16bf;
typedef __attribute__((ext_vector_type(8)))  float  v8f;

#define N_ROWS      32768
#define INTER       768
#define OUT_DIM     256
#define NNZ_PER_ROW 32
#define ROWS_PER_WG 32
#define KSTEPS      (INTER / 32)              // 24 WMMA K-steps
#define ACCF_STRIDE 772                       // f32 elems (768 + 4 pad)
#define ACCF_BYTES  (ROWS_PER_WG * ACCF_STRIDE * 4)   // 98816 B
#define NFRAGS      (KSTEPS * 2)              // 48 fragments (24 kb x 2 row-blocks)
#define FRAG_BYTES  (NFRAGS * 32 * 32)        // 49152 B (32 lanes x 32 B each)

__device__ __forceinline__ uint32_t f32_to_bf16_rne(float x) {
    uint32_t u = __builtin_bit_cast(uint32_t, x);
    uint32_t r = u + 0x7FFFu + ((u >> 16) & 1u);   // round-to-nearest-even
    return r >> 16;
}

// ---------------------------------------------------------------------------
// Kernel 1: W [768,256] f32 (row-major) -> Wt [256,768] bf16 (row-major).
// Transposed so each lane's B-fragment is two 16-byte loads from one Wt row.
// ---------------------------------------------------------------------------
__global__ void wt_convert_kernel(const float* __restrict__ W,
                                  uint16_t* __restrict__ Wt) {
    int g = blockIdx.x * blockDim.x + threadIdx.x;   // 0 .. 256*768-1
    int n = g / INTER;
    int k = g - n * INTER;
    Wt[g] = (uint16_t)f32_to_bf16_rne(W[k * OUT_DIM + n]);
}

// ---------------------------------------------------------------------------
// Kernel 2: fused scatter + GEMM. One WG (256 thr / 8 waves) per 32 rows.
// Dynamic LDS: f32 accumulator (96.5 KB) + bf16 fragment buffer (48 KB)
// = 144.5 KB -> two workgroups per 320 KB WGP.
// ---------------------------------------------------------------------------
__global__ __launch_bounds__(256)
void nnue_factored_kernel(const float* __restrict__ values,
                          const int*   __restrict__ batch_idx,
                          const int*   __restrict__ active_idx,
                          const int*   __restrict__ f,
                          const uint16_t* __restrict__ Wt,
                          float* __restrict__ out) {
    extern __shared__ char smem[];
    float*    accF    = (float*)smem;                       // [32][772] f32
    uint32_t* fragU32 = (uint32_t*)(smem + ACCF_BYTES);     // frag-layout bf16 tile
    uint64_t* fragU64 = (uint64_t*)(smem + ACCF_BYTES);

    const int tid     = threadIdx.x;
    const int rowBase = blockIdx.x * ROWS_PER_WG;
    const int nnzBase = rowBase * NNZ_PER_ROW;

    // ---- 1) zero the f32 accumulator tile -------------------------------
    for (int i = tid; i < ROWS_PER_WG * ACCF_STRIDE; i += 256)
        accF[i] = 0.0f;
    __syncthreads();

    // ---- 2) scatter-add this tile's 1024 nonzeros (ds_add_f32) ----------
    for (int i = tid; i < ROWS_PER_WG * NNZ_PER_ROW; i += 256) {
        int j = nnzBase + i;
        int r = batch_idx[j] - rowBase;        // rows are sorted: r in [0,32)
        int c = f[active_idx[j]];              // factored bucket in [0,768)
        atomicAdd(&accF[r * ACCF_STRIDE + c], values[j]);
    }
    __syncthreads();

    // ---- 3) convert to bf16 in WMMA A-fragment order --------------------
    // 16-bit A 16x32 layout: lane = m + 16*hi; VGPR j<4 holds K={2j,2j+1}+8*hi,
    // j>=4 holds K={16+2(j-4),17+2(j-4)}+8*hi. Fragment buffer interleave:
    // u64 slot index = fragId*128 + (j>>1)*32 + lane  (fragId = kb*2 + rb),
    // so each ds_load_b64 in phase 4 touches all 64 banks exactly once.
    for (int e2 = tid; e2 < ROWS_PER_WG * (INTER / 2); e2 += 256) {
        int r  = e2 / (INTER / 2);
        int cp = e2 - r * (INTER / 2);         // u32-pair column, 0..383
        int kb = cp >> 4;                      // K window
        int q  = cp & 15;                      // k_local >> 1
        int hi = (q >> 2) & 1;
        int j  = (q & 3) + ((q >> 3) << 2);    // A-fragment u32 slot
        int rb = r >> 4;
        int m  = r & 15;
        float lo = accF[r * ACCF_STRIDE + 2 * cp];
        float hg = accF[r * ACCF_STRIDE + 2 * cp + 1];
        uint32_t packed = f32_to_bf16_rne(lo) | (f32_to_bf16_rne(hg) << 16);
        fragU32[(kb * 2 + rb) * 256 + (j >> 1) * 64 + (m + 16 * hi) * 2 + (j & 1)]
            = packed;
    }
    __syncthreads();

    // ---- 4) WMMA: out[32 x 256] = tile[32 x 768] @ Wt^T -----------------
    // Each wave: 2 col-blocks x 2 row-blocks = 4 independent accumulators.
    const int wave   = tid >> 5;
    const int lane   = tid & 31;
    const int laneM  = lane & 15;
    const int laneHi = lane >> 4;
    const int cb0    = wave * 2;

    // B-fragment bases: transposed bf16 W, row n holds K contiguous;
    // lanes 0-15 take K=0..7,16..23, lanes 16-31 take K=8..15,24..31 of each window.
    const uint16_t* b0 = Wt + (cb0 * 16 + laneM) * INTER + laneHi * 8;
    const uint16_t* b1 = b0 + 16 * INTER;
    const uint64_t* afrag = fragU64 + lane;

    v8f accA0B0 = {}, accA0B1 = {}, accA1B0 = {}, accA1B1 = {};

    for (int kb = 0; kb < KSTEPS; ++kb) {
        // B fragments: two 16-byte per-lane loads each, 32 bytes apart.
        union { v16bf v; uint4 q[2]; } B0, B1;
        const char* bp0 = (const char*)(b0 + kb * 32);
        const char* bp1 = (const char*)(b1 + kb * 32);
        B0.q[0] = *(const uint4*)(bp0);
        B0.q[1] = *(const uint4*)(bp0 + 32);
        B1.q[0] = *(const uint4*)(bp1);
        B1.q[1] = *(const uint4*)(bp1 + 32);
        if (kb + 1 < KSTEPS) {
            __builtin_prefetch(bp0 + 64, 0, 3);   // next K panel
            __builtin_prefetch(bp1 + 64, 0, 3);
        }

        // A fragments: 4 conflict-free ds_load_b64, immediate offsets only.
        union { v16bf v; uint64_t q[4]; } A0, A1;
        const uint64_t* a0 = afrag + (kb * 2 + 0) * 128;
        const uint64_t* a1 = afrag + (kb * 2 + 1) * 128;
#pragma unroll
        for (int q64 = 0; q64 < 4; ++q64) {
            A0.q[q64] = a0[q64 * 32];
            A1.q[q64] = a1[q64 * 32];
        }

        // 4 independent WMMAs (no D->A/B hazards inside a K-step).
        accA0B0 = __builtin_amdgcn_wmma_f32_16x16x32_bf16(
                      false, A0.v, false, B0.v, (short)0, accA0B0, false, false);
        accA0B1 = __builtin_amdgcn_wmma_f32_16x16x32_bf16(
                      false, A0.v, false, B1.v, (short)0, accA0B1, false, false);
        accA1B0 = __builtin_amdgcn_wmma_f32_16x16x32_bf16(
                      false, A1.v, false, B0.v, (short)0, accA1B0, false, false);
        accA1B1 = __builtin_amdgcn_wmma_f32_16x16x32_bf16(
                      false, A1.v, false, B1.v, (short)0, accA1B1, false, false);
    }

    // ---- store C fragments per documented layout:
    // lane<16: (m=j, n=lane) ; lane>=16: (m=8+j, n=lane-16)
    const int n0 = cb0 * 16 + laneM;
    const int n1 = n0 + 16;
#pragma unroll
    for (int j = 0; j < 8; ++j) {
        int m = j + 8 * laneHi;
        float* o0 = out + (size_t)(rowBase +      m) * OUT_DIM;
        float* o1 = out + (size_t)(rowBase + 16 + m) * OUT_DIM;
        o0[n0] = accA0B0[j];
        o0[n1] = accA0B1[j];
        o1[n0] = accA1B0[j];
        o1[n1] = accA1B1[j];
    }
}

// ---------------------------------------------------------------------------
extern "C" void kernel_launch(void* const* d_in, const int* in_sizes, int n_in,
                              void* d_out, int out_size, void* d_ws, size_t ws_size,
                              hipStream_t stream) {
    (void)in_sizes; (void)n_in; (void)out_size; (void)ws_size;

    // setup_inputs() dict order: values, weights, batch_idx, active_idx, f
    const float* values     = (const float*)d_in[0];
    const float* weights    = (const float*)d_in[1];
    const int*   batch_idx  = (const int*)d_in[2];
    const int*   active_idx = (const int*)d_in[3];
    const int*   f          = (const int*)d_in[4];
    float*       out        = (float*)d_out;

    // d_ws: bf16 transposed weights, 256*768*2 = 384 KB (L2-resident).
    uint16_t* Wt = (uint16_t*)d_ws;

    wt_convert_kernel<<<(OUT_DIM * INTER) / 256, 256, 0, stream>>>(weights, Wt);

    const size_t smem = (size_t)ACCF_BYTES + (size_t)FRAG_BYTES;  // 144.5 KB
    nnue_factored_kernel<<<N_ROWS / ROWS_PER_WG, 256, smem, stream>>>(
        values, batch_idx, active_idx, f, Wt, out);
}
